// TrafficTransformerDecoder_3762391351340
// MI455X (gfx1250) — compile-verified, matrix operand order checked by reading, not run on hardware
//
#include <hip/hip_runtime.h>

// ---------------------------------------------------------------------------
// CDNA5 (gfx1250) wave32 WMMA implementation of the traffic transformer
// decoder. All large GEMMs run through V_WMMA_F32_16X16X32_BF16 with an
// LDS-staged, fragment-swizzled B operand; softmax, layernorm and the IDW
// adjacency stay in fp32 VALU + LDS.
// ---------------------------------------------------------------------------

typedef __bf16 bf16;
typedef __attribute__((ext_vector_type(16))) __bf16 v16bf;
typedef __attribute__((ext_vector_type(8)))  __bf16 v8bf;
typedef __attribute__((ext_vector_type(8)))  float  v8f;

#define B_      32
#define N_      64
#define F_IN_   32
#define G_      224
#define T_OUT_  24
#define T_IN_   168
#define NH_     8
#define HD_     32
#define D_      256
#define EF_     4
#define EPS_    1e-8f

#define BN_       (B_ * N_)          /* 2048   */
#define ROWS_DEC_ (BN_ * T_OUT_)     /* 49152  */
#define ROWS_ENC_ (BN_ * T_IN_)      /* 344064 */

__device__ __forceinline__ float b2f(bf16 v) { return (float)v; }
__device__ __forceinline__ bf16  f2b(float v) { return (bf16)v; }

// ---------------------------------------------------------------------------
// fp32 -> bf16 staging, optionally fusing a (B, Tlen, D) periodic embedding
// broadcast over the N stations.  pe == nullptr -> plain convert (weights).
// ---------------------------------------------------------------------------
__global__ void k_f2bf(const float* __restrict__ src, bf16* __restrict__ dst,
                       long n, const float* __restrict__ pe, int Tlen)
{
    long i = (long)blockIdx.x * blockDim.x + threadIdx.x;
    if (i >= n) return;
    float v = src[i];
    if (pe) {
        long d   = i & (D_ - 1);
        long row = i >> 8;                     // (b*N + n)*Tlen + t
        long t   = row % Tlen;
        long bb  = row / ((long)N_ * Tlen);
        v += pe[(bb * Tlen + t) * D_ + d];
    }
    dst[i] = f2b(v);
}

// ---------------------------------------------------------------------------
// IDW adjacency: row-normalized inverse-distance weights + self loops,
// then symmetric degree normalization.  Two passes (dis must be complete).
// ---------------------------------------------------------------------------
__global__ void k_deg(const float* __restrict__ locs, float* __restrict__ a,
                      float* __restrict__ dis)
{
    int row = blockIdx.x;              // b*N + i
    int b = row / N_, i = row - b * N_;
    int j = threadIdx.x;               // 0..63
    float xi = locs[(b * N_ + i) * 2 + 0], yi = locs[(b * N_ + i) * 2 + 1];
    float xj = locs[(b * N_ + j) * 2 + 0], yj = locs[(b * N_ + j) * 2 + 1];
    float dx = xi - xj, dy = yi - yj;
    float dist = sqrtf(dx * dx + dy * dy + EPS_);
    float w = (i == j) ? 0.0f : 1.0f / (dist + EPS_);

    __shared__ float red[N_];
    red[j] = w;
    __syncthreads();
    for (int s = 32; s > 0; s >>= 1) {
        if (j < s) red[j] += red[j + s];
        __syncthreads();
    }
    float sum = red[0];
    float wn = w / (sum + EPS_);
    a[(long)row * N_ + j] = wn + (i == j ? 1.0f : 0.0f);
    if (j == 0) {
        float rowsum_a = sum / (sum + EPS_) + 1.0f;   // sum_j wn + self loop
        dis[row] = rsqrtf(rowsum_a + EPS_);
    }
}

__global__ void k_anorm(const float* __restrict__ a, const float* __restrict__ dis,
                        bf16* __restrict__ out)
{
    int row = blockIdx.x;
    int j = threadIdx.x;
    int b = row / N_;
    float v = dis[row] * a[(long)row * N_ + j] * dis[b * N_ + j];
    out[(long)row * N_ + j] = f2b(v);
}

// ---------------------------------------------------------------------------
// Block-tiled batched bf16 WMMA GEMM:  C[M,Nc] = A[M,K]*B[K,Nc] (+bias)(+relu)
//   block = 128 threads (4 waves) -> 64x32 C macro-tile
//   wave  = 16 rows x 2 column tiles (A fragment reused across 2 WMMAs)
//   B 32x32 K-slab loaded once per block with vector global_load_b128 and
//     scattered into WMMA fragment order in LDS (ds_load_b128 at use)
// Requires M % 64 == 0, Nc % 32 == 0, K % 32 == 0 (true for every call here),
// so no edge handling and EXEC stays all-ones for WMMA.
// ---------------------------------------------------------------------------
template <int RELU, int BF16OUT>
__global__ __launch_bounds__(128)
void k_gemm(const bf16* __restrict__ A, const bf16* __restrict__ Bm,
            const float* __restrict__ bias,
            float* __restrict__ Cf, bf16* __restrict__ Cb,
            int M, int Nc, int K,
            long strideA, long strideB, long strideC,
            int tilesM64, int tilesN32)
{
    __shared__ v16bf bfrag[2][32];     // [n-subtile][lane] fragment-order B slab

    int blk = blockIdx.x;
    int perBatch = tilesM64 * tilesN32;
    int batch = blk / perBatch;
    int t = blk - batch * perBatch;
    int tm = t / tilesN32;
    int tn = t - tm * tilesN32;

    const bf16* Ab = A + (long)batch * strideA;
    const bf16* Bb = Bm + (long)batch * strideB;

    int tid  = threadIdx.x;
    int wv   = tid >> 5;
    int lane = tid & 31;
    int half = lane >> 4;     // K-octet selector
    int r    = lane & 15;
    int rowA = tm * 64 + wv * 16 + r;  // A row this lane supplies
    int colN = tn * 32;                 // macro-tile first column

    // ---- fill mapping: thread -> (B row k, 8-column chunk) ----------------
    int fk    = tid >> 2;          // 0..31 : K row within slab
    int chunk = tid & 3;           // 0..3  : 8-column chunk
    int fhw = (fk >> 3) & 1;
    int flo = fk & 1;
    int fi  = ((fk < 16) ? 0 : 4) + ((fk & 7) >> 1);
    int fe  = 2 * fi + flo;        // fragment element index for this B row

    v8f acc0 = {}, acc1 = {};
    for (int k0 = 0; k0 < K; k0 += 32) {
        // ---- cooperative B slab: one b128 vector load + LDS scatter -------
        __syncthreads();
        {
            v8bf vals = *(const v8bf*)(Bb + (long)(k0 + fk) * Nc + colN + chunk * 8);
#pragma unroll
            for (int j = 0; j < 8; ++j) {
                int col  = chunk * 8 + j;
                int tile = col >> 4;
                int lw   = fhw * 16 + (col & 15);
                ((bf16*)&bfrag[tile][lw])[fe] = vals[j];
            }
        }
        __syncthreads();

        // ---- A fragment (direct global, vectorizes to global_load_b128) ---
        if (k0 + 32 < K)
            __builtin_prefetch(Ab + (long)rowA * K + k0 + 32, 0, 1);
        v16bf af;
#pragma unroll
        for (int i = 0; i < 8; ++i) {
            int kb = ((i < 4) ? 2 * i : 16 + 2 * (i - 4)) + half * 8 + k0;
            af[2 * i]     = Ab[(long)rowA * K + kb];
            af[2 * i + 1] = Ab[(long)rowA * K + kb + 1];
        }

        // ---- two back-to-back WMMAs reusing the A fragment ----------------
        v16bf b0 = bfrag[0][lane];
        v16bf b1 = bfrag[1][lane];
        acc0 = __builtin_amdgcn_wmma_f32_16x16x32_bf16(
            false, af, false, b0, (short)0, acc0, false, false);
        acc1 = __builtin_amdgcn_wmma_f32_16x16x32_bf16(
            false, af, false, b1, (short)0, acc1, false, false);
    }

    long cbase = (long)batch * strideC;
    float bb0 = bias ? bias[colN + r]      : 0.0f;
    float bb1 = bias ? bias[colN + 16 + r] : 0.0f;
#pragma unroll
    for (int i = 0; i < 8; ++i) {
        int rowC = tm * 64 + wv * 16 + half * 8 + i;  // C/D: VGPR i -> M = i+8*half
        float v0 = acc0[i] + bb0;
        float v1 = acc1[i] + bb1;
        if (RELU) { v0 = v0 > 0.0f ? v0 : 0.0f; v1 = v1 > 0.0f ? v1 : 0.0f; }
        long i0 = cbase + (long)rowC * Nc + colN + r;
        long i1 = i0 + 16;
        if (BF16OUT) { Cb[i0] = f2b(v0); Cb[i1] = f2b(v1); }
        else         { Cf[i0] = v0;      Cf[i1] = v1; }
    }
}

// ---------------------------------------------------------------------------
// Concatenate [dec_in | gcn2] and add positional embedding -> x (fp32).
// ---------------------------------------------------------------------------
__global__ void k_build_x(const float* __restrict__ dec_in,
                          const float* __restrict__ gcn2,
                          const float* __restrict__ pos,
                          float* __restrict__ x)
{
    long i = (long)blockIdx.x * blockDim.x + threadIdx.x;
    if (i >= (long)ROWS_DEC_ * D_) return;
    long d = i & (D_ - 1);
    long row = i >> 8;
    long t = row % T_OUT_;
    long bb = row / ((long)N_ * T_OUT_);
    float v = (d < F_IN_) ? dec_in[row * F_IN_ + d]
                          : gcn2[row * G_ + (d - F_IN_)];
    v += pos[(bb * T_OUT_ + t) * D_ + d];
    x[i] = v;
}

// ---------------------------------------------------------------------------
// Self-attention: one (bn, head) per block, 24x24 scores, LDS-resident.
// ---------------------------------------------------------------------------
__global__ void k_self_attn(const bf16* __restrict__ Q, const bf16* __restrict__ K,
                            const bf16* __restrict__ V, bf16* __restrict__ O)
{
    int blk = blockIdx.x;
    int bn = blk / NH_, h = blk - bn * NH_;
    __shared__ float sq[T_OUT_][HD_], sk[T_OUT_][HD_], sv[T_OUT_][HD_];
    int tid = threadIdx.x;   // 32
    const long base = (long)bn * T_OUT_ * D_ + h * HD_;
    for (int idx = tid; idx < T_OUT_ * HD_; idx += 32) {
        int t = idx / HD_, d = idx - (idx / HD_) * HD_;
        sq[t][d] = b2f(Q[base + (long)t * D_ + d]);
        sk[t][d] = b2f(K[base + (long)t * D_ + d]);
        sv[t][d] = b2f(V[base + (long)t * D_ + d]);
    }
    __syncthreads();
    if (tid < T_OUT_) {
        const float scale = 0.17677669529663687f;  // 1/sqrt(32)
        float sc[T_OUT_], mx = -1e30f;
        for (int k = 0; k < T_OUT_; ++k) {
            float s = 0.0f;
            for (int d = 0; d < HD_; ++d) s += sq[tid][d] * sk[k][d];
            s *= scale;
            sc[k] = s;
            mx = fmaxf(mx, s);
        }
        float sum = 0.0f;
        for (int k = 0; k < T_OUT_; ++k) { sc[k] = __expf(sc[k] - mx); sum += sc[k]; }
        float inv = 1.0f / sum;
        float o[HD_];
        for (int d = 0; d < HD_; ++d) o[d] = 0.0f;
        for (int k = 0; k < T_OUT_; ++k) {
            float p = sc[k] * inv;
            for (int d = 0; d < HD_; ++d) o[d] += p * sv[k][d];
        }
        for (int d = 0; d < HD_; ++d)
            O[base + (long)tid * D_ + d] = f2b(o[d]);
    }
}

// ---------------------------------------------------------------------------
// Cross-attention: 24 queries x 168 keys, streaming (online) softmax.
// ---------------------------------------------------------------------------
__global__ void k_cross_attn(const bf16* __restrict__ Q, const bf16* __restrict__ K,
                             const bf16* __restrict__ V, bf16* __restrict__ O)
{
    int blk = blockIdx.x;
    int bn = blk / NH_, h = blk - bn * NH_;
    __shared__ float sq[T_OUT_][HD_];
    __shared__ float sk[T_IN_][HD_];
    __shared__ float sv[T_IN_][HD_];
    int tid = threadIdx.x;   // 64
    const long qbase = (long)bn * T_OUT_ * D_ + h * HD_;
    const long kbase = (long)bn * T_IN_ * D_ + h * HD_;
    for (int idx = tid; idx < T_OUT_ * HD_; idx += 64) {
        int t = idx / HD_, d = idx - (idx / HD_) * HD_;
        sq[t][d] = b2f(Q[qbase + (long)t * D_ + d]);
    }
    for (int idx = tid; idx < T_IN_ * HD_; idx += 64) {
        int t = idx / HD_, d = idx - (idx / HD_) * HD_;
        sk[t][d] = b2f(K[kbase + (long)t * D_ + d]);
        sv[t][d] = b2f(V[kbase + (long)t * D_ + d]);
    }
    __syncthreads();
    if (tid < T_OUT_) {
        const float scale = 0.17677669529663687f;
        float m = -1e30f, s = 0.0f;
        float o[HD_];
        for (int d = 0; d < HD_; ++d) o[d] = 0.0f;
        for (int k = 0; k < T_IN_; ++k) {
            float x = 0.0f;
            for (int d = 0; d < HD_; ++d) x += sq[tid][d] * sk[k][d];
            x *= scale;
            if (x > m) {
                float c = __expf(m - x);
                s *= c;
                for (int d = 0; d < HD_; ++d) o[d] *= c;
                m = x;
            }
            float p = __expf(x - m);
            s += p;
            for (int d = 0; d < HD_; ++d) o[d] += p * sv[k][d];
        }
        float inv = 1.0f / s;
        for (int d = 0; d < HD_; ++d)
            O[qbase + (long)tid * D_ + d] = f2b(o[d] * inv);
    }
}

// ---------------------------------------------------------------------------
// out = LayerNorm(X + Y) over last dim (D_ == blockDim.x == 256).
// ---------------------------------------------------------------------------
__global__ void k_add_ln(const float* __restrict__ X, const float* __restrict__ Y,
                         const float* __restrict__ g, const float* __restrict__ bta,
                         float* __restrict__ out)
{
    int row = blockIdx.x;
    int d = threadIdx.x;
    float v = X[(long)row * D_ + d] + Y[(long)row * D_ + d];
    __shared__ float red[D_];
    red[d] = v;
    __syncthreads();
    for (int s = 128; s > 0; s >>= 1) { if (d < s) red[d] += red[d + s]; __syncthreads(); }
    float mean = red[0] * (1.0f / D_);
    __syncthreads();
    float c = v - mean;
    red[d] = c * c;
    __syncthreads();
    for (int s = 128; s > 0; s >>= 1) { if (d < s) red[d] += red[d + s]; __syncthreads(); }
    float var = red[0] * (1.0f / D_);
    out[(long)row * D_ + d] = c * rsqrtf(var + 1e-5f) * g[d] + bta[d];
}

// ---------------------------------------------------------------------------
// Host side
// ---------------------------------------------------------------------------
static void launch_gemm(hipStream_t s, const bf16* A, const bf16* Bm,
                        const float* bias, float* Cf, bf16* Cb,
                        int M, int Nc, int K,
                        long sA, long sB, long sC, int nb, bool relu)
{
    int tm = M / 64, tn = Nc / 32;
    int blocks = tm * tn * nb;
    dim3 g(blocks), b(128);
    if (relu) {
        if (Cb) k_gemm<1, 1><<<g, b, 0, s>>>(A, Bm, bias, Cf, Cb, M, Nc, K, sA, sB, sC, tm, tn);
        else    k_gemm<1, 0><<<g, b, 0, s>>>(A, Bm, bias, Cf, Cb, M, Nc, K, sA, sB, sC, tm, tn);
    } else {
        if (Cb) k_gemm<0, 1><<<g, b, 0, s>>>(A, Bm, bias, Cf, Cb, M, Nc, K, sA, sB, sC, tm, tn);
        else    k_gemm<0, 0><<<g, b, 0, s>>>(A, Bm, bias, Cf, Cb, M, Nc, K, sA, sB, sC, tm, tn);
    }
}

static void conv_bf16(hipStream_t s, const float* src, bf16* dst, long n,
                      const float* pe, int Tlen)
{
    int blocks = (int)((n + 255) / 256);
    k_f2bf<<<blocks, 256, 0, s>>>(src, dst, n, pe, Tlen);
}

extern "C" void kernel_launch(void* const* d_in, const int* in_sizes, int n_in,
                              void* d_out, int out_size, void* d_ws, size_t ws_size,
                              hipStream_t stream)
{
    (void)in_sizes; (void)n_in; (void)out_size; (void)ws_size;
    const float* dec_in  = (const float*)d_in[0];
    const float* enc_out = (const float*)d_in[1];
    const float* locs    = (const float*)d_in[2];
    const float* dec_pos = (const float*)d_in[3];
    const float* enc_pe  = (const float*)d_in[4];
    const float* dec_pe  = (const float*)d_in[5];
    const float* gcn_w = (const float*)d_in[6];
    const float* gcn_b = (const float*)d_in[7];
    const float* fc1_w = (const float*)d_in[8];
    const float* fc1_b = (const float*)d_in[9];
    const float* ff_w1 = (const float*)d_in[10];
    const float* ff_b1 = (const float*)d_in[11];
    const float* ff_w2 = (const float*)d_in[12];
    const float* ff_b2 = (const float*)d_in[13];
    const float* ln1_g = (const float*)d_in[14];
    const float* ln1_b = (const float*)d_in[15];
    const float* ln2_g = (const float*)d_in[16];
    const float* ln2_b = (const float*)d_in[17];
    const float* ln3_g = (const float*)d_in[18];
    const float* ln3_b = (const float*)d_in[19];
    const float* sa_wq = (const float*)d_in[20];
    const float* sa_bq = (const float*)d_in[21];
    const float* sa_wk = (const float*)d_in[22];
    const float* sa_bk = (const float*)d_in[23];
    const float* sa_wv = (const float*)d_in[24];
    const float* sa_bv = (const float*)d_in[25];
    const float* sa_wo = (const float*)d_in[26];
    const float* sa_bo = (const float*)d_in[27];
    const float* ca_wq = (const float*)d_in[28];
    const float* ca_bq = (const float*)d_in[29];
    const float* ca_wk = (const float*)d_in[30];
    const float* ca_bk = (const float*)d_in[31];
    const float* ca_wv = (const float*)d_in[32];
    const float* ca_bv = (const float*)d_in[33];
    const float* ca_wo = (const float*)d_in[34];
    const float* ca_bo = (const float*)d_in[35];

    // ---- workspace bump allocator ------------------------------------------
    size_t off = 0;
    auto alloc = [&](size_t bytes) -> char* {
        char* p = (char*)d_ws + off;
        off += (bytes + 255) & ~(size_t)255;
        return p;
    };

    float* dis      = (float*)alloc((size_t)BN_ * 4);
    float* a_f      = (float*)alloc((size_t)B_ * N_ * N_ * 4);
    bf16*  anorm_bf = (bf16*) alloc((size_t)B_ * N_ * N_ * 2);
    bf16*  decin_bf = (bf16*) alloc((size_t)ROWS_DEC_ * F_IN_ * 2);
    float* agg_f    = (float*)alloc((size_t)ROWS_DEC_ * F_IN_ * 4);
    bf16*  agg_bf   = (bf16*) alloc((size_t)ROWS_DEC_ * F_IN_ * 2);

    bf16* w_gcn = (bf16*)alloc((size_t)F_IN_ * G_ * 2);
    bf16* w_fc1 = (bf16*)alloc((size_t)G_ * G_ * 2);
    bf16* w_saq = (bf16*)alloc((size_t)D_ * D_ * 2);
    bf16* w_sak = (bf16*)alloc((size_t)D_ * D_ * 2);
    bf16* w_sav = (bf16*)alloc((size_t)D_ * D_ * 2);
    bf16* w_sao = (bf16*)alloc((size_t)D_ * D_ * 2);
    bf16* w_caq = (bf16*)alloc((size_t)D_ * D_ * 2);
    bf16* w_cak = (bf16*)alloc((size_t)D_ * D_ * 2);
    bf16* w_cav = (bf16*)alloc((size_t)D_ * D_ * 2);
    bf16* w_cao = (bf16*)alloc((size_t)D_ * D_ * 2);
    bf16* w_ff1 = (bf16*)alloc((size_t)D_ * EF_ * D_ * 2);
    bf16* w_ff2 = (bf16*)alloc((size_t)EF_ * D_ * D_ * 2);

    bf16*  gcn1_bf = (bf16*) alloc((size_t)ROWS_DEC_ * G_ * 2);
    float* gcn2_f  = (float*)alloc((size_t)ROWS_DEC_ * G_ * 4);

    float* bufA = (float*)alloc((size_t)ROWS_DEC_ * D_ * 4);  // x   -> x2
    float* bufB = (float*)alloc((size_t)ROWS_DEC_ * D_ * 4);  // sa  -> ca -> ff
    float* bufC = (float*)alloc((size_t)ROWS_DEC_ * D_ * 4);  // x1

    bf16* act_q  = (bf16*)alloc((size_t)ROWS_DEC_ * D_ * 2);  // x+pe   / x1+pe
    bf16* act_v  = (bf16*)alloc((size_t)ROWS_DEC_ * D_ * 2);  // x      / x2
    bf16* act_Q  = (bf16*)alloc((size_t)ROWS_DEC_ * D_ * 2);  // Q      / Qc
    bf16* act_K  = (bf16*)alloc((size_t)ROWS_DEC_ * D_ * 2);
    bf16* act_V  = (bf16*)alloc((size_t)ROWS_DEC_ * D_ * 2);
    bf16* act_At = (bf16*)alloc((size_t)ROWS_DEC_ * D_ * 2);  // attn out (both)

    bf16* memk_bf = (bf16*)alloc((size_t)ROWS_ENC_ * D_ * 2);
    bf16* memv_bf = (bf16*)alloc((size_t)ROWS_ENC_ * D_ * 2);
    bf16* kc_bf   = (bf16*)alloc((size_t)ROWS_ENC_ * D_ * 2);
    bf16* vc_bf   = (bf16*)alloc((size_t)ROWS_ENC_ * D_ * 2);
    bf16* h_bf    = (bf16*)alloc((size_t)ROWS_DEC_ * EF_ * D_ * 2);

    // ---- weights -> bf16 ----------------------------------------------------
    conv_bf16(stream, gcn_w, w_gcn, (long)F_IN_ * G_, nullptr, 0);
    conv_bf16(stream, fc1_w, w_fc1, (long)G_ * G_, nullptr, 0);
    conv_bf16(stream, sa_wq, w_saq, (long)D_ * D_, nullptr, 0);
    conv_bf16(stream, sa_wk, w_sak, (long)D_ * D_, nullptr, 0);
    conv_bf16(stream, sa_wv, w_sav, (long)D_ * D_, nullptr, 0);
    conv_bf16(stream, sa_wo, w_sao, (long)D_ * D_, nullptr, 0);
    conv_bf16(stream, ca_wq, w_caq, (long)D_ * D_, nullptr, 0);
    conv_bf16(stream, ca_wk, w_cak, (long)D_ * D_, nullptr, 0);
    conv_bf16(stream, ca_wv, w_cav, (long)D_ * D_, nullptr, 0);
    conv_bf16(stream, ca_wo, w_cao, (long)D_ * D_, nullptr, 0);
    conv_bf16(stream, ff_w1, w_ff1, (long)D_ * EF_ * D_, nullptr, 0);
    conv_bf16(stream, ff_w2, w_ff2, (long)EF_ * D_ * D_, nullptr, 0);

    // ---- GCN ---------------------------------------------------------------
    k_deg<<<BN_, N_, 0, stream>>>(locs, a_f, dis);
    k_anorm<<<BN_, N_, 0, stream>>>(a_f, dis, anorm_bf);
    conv_bf16(stream, dec_in, decin_bf, (long)ROWS_DEC_ * F_IN_, nullptr, 0);
    // agg[b] = a_norm[b] (64x64) @ dec_in[b] (64x768), batched over B
    launch_gemm(stream, anorm_bf, decin_bf, nullptr, agg_f, nullptr,
                N_, T_OUT_ * F_IN_, N_,
                (long)N_ * N_, (long)N_ * T_OUT_ * F_IN_, (long)N_ * T_OUT_ * F_IN_,
                B_, false);
    conv_bf16(stream, agg_f, agg_bf, (long)ROWS_DEC_ * F_IN_, nullptr, 0);
    launch_gemm(stream, agg_bf, w_gcn, gcn_b, nullptr, gcn1_bf,
                ROWS_DEC_, G_, F_IN_, 0, 0, 0, 1, true);          // relu
    launch_gemm(stream, gcn1_bf, w_fc1, fc1_b, gcn2_f, nullptr,
                ROWS_DEC_, G_, G_, 0, 0, 0, 1, false);

    // ---- x = [dec_in | gcn] + pos_em --------------------------------------
    {
        long n = (long)ROWS_DEC_ * D_;
        k_build_x<<<(int)((n + 255) / 256), 256, 0, stream>>>(dec_in, gcn2_f, dec_pos, bufA);
    }
    conv_bf16(stream, bufA, act_q, (long)ROWS_DEC_ * D_, dec_pe, T_OUT_);  // x + dec_pe
    conv_bf16(stream, bufA, act_v, (long)ROWS_DEC_ * D_, nullptr, 0);      // x

    // ---- self-attention ----------------------------------------------------
    launch_gemm(stream, act_q, w_saq, sa_bq, nullptr, act_Q, ROWS_DEC_, D_, D_, 0, 0, 0, 1, false);
    launch_gemm(stream, act_q, w_sak, sa_bk, nullptr, act_K, ROWS_DEC_, D_, D_, 0, 0, 0, 1, false);
    launch_gemm(stream, act_v, w_sav, sa_bv, nullptr, act_V, ROWS_DEC_, D_, D_, 0, 0, 0, 1, false);
    k_self_attn<<<BN_ * NH_, 32, 0, stream>>>(act_Q, act_K, act_V, act_At);
    launch_gemm(stream, act_At, w_sao, sa_bo, bufB, nullptr, ROWS_DEC_, D_, D_, 0, 0, 0, 1, false);
    k_add_ln<<<ROWS_DEC_, D_, 0, stream>>>(bufA, bufB, ln1_g, ln1_b, bufC); // x1

    // ---- cross-attention ---------------------------------------------------
    conv_bf16(stream, bufC, act_q, (long)ROWS_DEC_ * D_, dec_pe, T_OUT_);      // x1 + dec_pe
    conv_bf16(stream, enc_out, memk_bf, (long)ROWS_ENC_ * D_, enc_pe, T_IN_);  // mem + enc_pe
    conv_bf16(stream, enc_out, memv_bf, (long)ROWS_ENC_ * D_, nullptr, 0);     // mem
    launch_gemm(stream, act_q,   w_caq, ca_bq, nullptr, act_Q, ROWS_DEC_, D_, D_, 0, 0, 0, 1, false);
    launch_gemm(stream, memk_bf, w_cak, ca_bk, nullptr, kc_bf, ROWS_ENC_, D_, D_, 0, 0, 0, 1, false);
    launch_gemm(stream, memv_bf, w_cav, ca_bv, nullptr, vc_bf, ROWS_ENC_, D_, D_, 0, 0, 0, 1, false);
    k_cross_attn<<<BN_ * NH_, 64, 0, stream>>>(act_Q, kc_bf, vc_bf, act_At);
    launch_gemm(stream, act_At, w_cao, ca_bo, bufB, nullptr, ROWS_DEC_, D_, D_, 0, 0, 0, 1, false);
    k_add_ln<<<ROWS_DEC_, D_, 0, stream>>>(bufC, bufB, ln2_g, ln2_b, bufA);  // x2

    // ---- feed-forward ------------------------------------------------------
    conv_bf16(stream, bufA, act_v, (long)ROWS_DEC_ * D_, nullptr, 0);
    launch_gemm(stream, act_v, w_ff1, ff_b1, nullptr, h_bf,
                ROWS_DEC_, EF_ * D_, D_, 0, 0, 0, 1, true);                  // relu
    launch_gemm(stream, h_bf, w_ff2, ff_b2, bufB, nullptr,
                ROWS_DEC_, D_, EF_ * D_, 0, 0, 0, 1, false);
    k_add_ln<<<ROWS_DEC_, D_, 0, stream>>>(bufA, bufB, ln3_g, ln3_b, (float*)d_out);
}
